// MultiScaleSpatialLatentAttention_4466765988089
// MI455X (gfx1250) — matrix operand order repaired and checked by reading, not compiled, not run on hardware
//
#include <hip/hip_runtime.h>
#include <hip/hip_bf16.h>
#include <math.h>

// ---------------------------------------------------------------------------
// MultiScaleSpatialLatentAttention for MI455X (gfx1250, wave32, WMMA)
// All GEMMs + attention run on v_wmma_f32_16x16x32_bf16 (f32 accumulate).
// ---------------------------------------------------------------------------

typedef __bf16 bf16_t;
typedef __attribute__((ext_vector_type(16))) __bf16 v16bf;
typedef __attribute__((ext_vector_type(8)))  __bf16 v8bf;
typedef __attribute__((ext_vector_type(8)))  float  v8f;

#define BB     16
#define TT     576
#define DMODEL 768
#define NHH    12
#define HDD    64
#define LDIM   256

__device__ __forceinline__ v8f vzero8() {
  v8f z;
#pragma unroll
  for (int i = 0; i < 8; ++i) z[i] = 0.0f;
  return z;
}

__device__ __forceinline__ v16bf frag_cat(v8bf lo, v8bf hi) {
  return __builtin_shufflevector(lo, hi, 0, 1, 2, 3, 4, 5, 6, 7,
                                         8, 9, 10, 11, 12, 13, 14, 15);
}

// A fragment: 16x32 bf16, rows contiguous along K (row-major M x K).
// Lane l: row = l&15, half = l>>4. VGPR0-3 hold K = half*8 + 0..7,
// VGPR4-7 hold K = 16 + half*8 + 0..7  -> two contiguous 16B loads.
__device__ __forceinline__ v16bf ld_frag_a(const bf16_t* base, int ld, int k0) {
  const int lane = threadIdx.x & 31;
  const int m = lane & 15, hi = lane >> 4;
  const bf16_t* p = base + (size_t)m * ld + k0 + hi * 8;
  v8bf lo = *(const v8bf*)p;
  v8bf hh = *(const v8bf*)(p + 16);
  return frag_cat(lo, hh);
}

// B fragment: 32x16 bf16 where operand is stored N x K row-major (K contig).
// Lane l: col = l&15, half = l>>4. VGPR0-3 hold K = half*8 + 0..7,
// VGPR4-7 hold K = 16 + half*8 + 0..7  -> two contiguous 16B loads.
__device__ __forceinline__ v16bf ld_frag_b(const bf16_t* base, int ld, int k0) {
  const int lane = threadIdx.x & 31;
  const int n = lane & 15, hi = lane >> 4;
  const bf16_t* p = base + (size_t)n * ld + k0 + hi * 8;
  v8bf lo = *(const v8bf*)p;
  v8bf hh = *(const v8bf*)(p + 16);
  return frag_cat(lo, hh);
}

__device__ __forceinline__ v8f wmma_bf16(v16bf a, v16bf b, v8f c) {
  return __builtin_amdgcn_wmma_f32_16x16x32_bf16(false, a, false, b,
                                                 (short)0, c, false, false);
}

// ---------------------------------------------------------------------------
// Generic WMMA GEMM: C[M,N] = A[M,K] * Bt[N,K]^T   (both bf16, K contiguous)
// One wave computes a 64x32 macro-tile (4 M-tiles x 2 N-tiles).
// Two-stage ping-pong pipeline (requires K % 64 == 0, true for K=256/768):
// buffer roles alternate structurally, so no register-rotation moves, and
// each WMMA group only waits on loads issued a full stage earlier.
// Cb != null  -> store bf16 ;  else store f32 (+ optional bias[n]).
// ---------------------------------------------------------------------------
__global__ __launch_bounds__(256) void gemm_bf16_wmma(
    const bf16_t* __restrict__ A, const bf16_t* __restrict__ Bt,
    bf16_t* __restrict__ Cb, float* __restrict__ Cf,
    const float* __restrict__ bias, int M, int N, int K)
{
  const int lane = threadIdx.x & 31;
  const int wid  = blockIdx.x * (blockDim.x >> 5) + (threadIdx.x >> 5);
  const int ntiles = N >> 5;               // 32-wide macro columns
  const int mtiles = M >> 6;               // 64-tall macro rows
  if (wid >= mtiles * ntiles) return;
  const int mt = wid / ntiles;
  const int nt = wid - mt * ntiles;
  const int m0 = mt << 6;
  const int n0 = nt << 5;
  const int n  = lane & 15;
  const int hi = lane >> 4;

  const bf16_t* arow[4];
#pragma unroll
  for (int t = 0; t < 4; ++t) arow[t] = A + (size_t)(m0 + 16 * t) * K;
  const bf16_t* brow[2];
#pragma unroll
  for (int u = 0; u < 2; ++u) brow[u] = Bt + (size_t)(n0 + 16 * u) * K;

  v8f acc[4][2];
#pragma unroll
  for (int t = 0; t < 4; ++t)
#pragma unroll
    for (int u = 0; u < 2; ++u) acc[t][u] = vzero8();

  // ---- prologue: stage-0 (k=0) and stage-1 (k=32) fragments ---------------
  v16bf a0[4], b0[2], a1[4], b1[2];
#pragma unroll
  for (int u = 0; u < 2; ++u) b0[u] = ld_frag_b(brow[u], K, 0);
#pragma unroll
  for (int t = 0; t < 4; ++t) a0[t] = ld_frag_a(arow[t], K, 0);
#pragma unroll
  for (int u = 0; u < 2; ++u) b1[u] = ld_frag_b(brow[u], K, 32);
#pragma unroll
  for (int t = 0; t < 4; ++t) a1[t] = ld_frag_a(arow[t], K, 32);

  for (int k0 = 64; k0 < K; k0 += 64) {
    if (k0 + 128 < K) {
      __builtin_prefetch(brow[0] + (size_t)n * K + k0 + 128, 0, 2);
      __builtin_prefetch(brow[1] + (size_t)n * K + k0 + 128, 0, 2);
      __builtin_prefetch(arow[0] + (size_t)n * K + k0 + 128, 0, 2);
      __builtin_prefetch(arow[2] + (size_t)n * K + k0 + 128, 0, 2);
    }
    // stage 0: consume buf0, refill buf0 at k0
#pragma unroll
    for (int t = 0; t < 4; ++t)
#pragma unroll
      for (int u = 0; u < 2; ++u)
        acc[t][u] = wmma_bf16(a0[t], b0[u], acc[t][u]);
#pragma unroll
    for (int u = 0; u < 2; ++u) b0[u] = ld_frag_b(brow[u], K, k0);
#pragma unroll
    for (int t = 0; t < 4; ++t) a0[t] = ld_frag_a(arow[t], K, k0);
    // stage 1: consume buf1, refill buf1 at k0+32
#pragma unroll
    for (int t = 0; t < 4; ++t)
#pragma unroll
      for (int u = 0; u < 2; ++u)
        acc[t][u] = wmma_bf16(a1[t], b1[u], acc[t][u]);
#pragma unroll
    for (int u = 0; u < 2; ++u) b1[u] = ld_frag_b(brow[u], K, k0 + 32);
#pragma unroll
    for (int t = 0; t < 4; ++t) a1[t] = ld_frag_a(arow[t], K, k0 + 32);
  }
  // ---- epilogue: drain both stages ----------------------------------------
#pragma unroll
  for (int t = 0; t < 4; ++t)
#pragma unroll
    for (int u = 0; u < 2; ++u)
      acc[t][u] = wmma_bf16(a0[t], b0[u], acc[t][u]);
#pragma unroll
  for (int t = 0; t < 4; ++t)
#pragma unroll
    for (int u = 0; u < 2; ++u)
      acc[t][u] = wmma_bf16(a1[t], b1[u], acc[t][u]);

#pragma unroll
  for (int u = 0; u < 2; ++u) {
    const int col = n0 + 16 * u + n;
    const float bv = bias ? bias[col] : 0.0f;
#pragma unroll
    for (int t = 0; t < 4; ++t) {
#pragma unroll
      for (int v = 0; v < 8; ++v) {
        const int r = m0 + 16 * t + v + 8 * hi;  // C layout: row = v + 8*half
        if (Cb) Cb[(size_t)r * N + col] = (bf16_t)acc[t][u][v];
        else    Cf[(size_t)r * N + col] = acc[t][u][v] + bv;
      }
    }
  }
}

// ---------------------------------------------------------------------------
// Elementwise / layout helper kernels
// ---------------------------------------------------------------------------
__global__ void f32_to_bf16_k(const float* __restrict__ in,
                              bf16_t* __restrict__ out, int nElem)
{
  int i = blockIdx.x * blockDim.x + threadIdx.x;
  if (i < nElem) out[i] = (bf16_t)in[i];
}

// W (K x N) f32  ->  Wt (N x K) bf16
__global__ void transpose_to_bf16_k(const float* __restrict__ W,
                                    bf16_t* __restrict__ Wt, int K, int N)
{
  int i = blockIdx.x * blockDim.x + threadIdx.x;
  if (i >= N * K) return;
  int nn = i / K, kk = i - nn * K;
  Wt[i] = (bf16_t)W[(size_t)kk * N + nn];
}

// lat (B*576, 256) f32  ->  lp (B*Tsp, 256) bf16, s x s mean pool, zero pads.
__global__ void pool_lat_k(const float* __restrict__ lat,
                           bf16_t* __restrict__ lp,
                           int s, int gw, int Ts, int Tsp)
{
  int i = blockIdx.x * blockDim.x + threadIdx.x;
  int total = BB * Tsp * LDIM;
  if (i >= total) return;
  int c   = i & (LDIM - 1);
  int row = i >> 8;
  int tok = row % Tsp;
  int b   = row / Tsp;
  if (tok >= Ts) { lp[i] = (bf16_t)0.0f; return; }
  int gi = tok / gw, gj = tok - gi * gw;
  float sum = 0.0f;
  for (int dy = 0; dy < s; ++dy)
    for (int dx = 0; dx < s; ++dx) {
      int srow = b * TT + (gi * s + dy) * 24 + (gj * s + dx);
      sum += lat[(size_t)srow * LDIM + c];
    }
  lp[i] = (bf16_t)(sum / (float)(s * s));
}

// Vg (B*Tsp, 768) bf16  ->  Vt (B, NH, 64, Tsp) bf16  (key index contiguous)
__global__ void v_transpose_k(const bf16_t* __restrict__ Vg,
                              bf16_t* __restrict__ Vt, int Tsp)
{
  int i = blockIdx.x * blockDim.x + threadIdx.x;
  int total = BB * NHH * HDD * Tsp;
  if (i >= total) return;
  int nn   = i % Tsp;
  int rest = i / Tsp;
  int d    = rest % HDD;
  int bh   = rest / HDD;
  int h = bh % NHH, b = bh / NHH;
  Vt[i] = Vg[(size_t)(b * Tsp + nn) * DMODEL + h * HDD + d];
}

// ---------------------------------------------------------------------------
// Multi-scale flash attention. One wave: 16 query rows of one (b, h).
// ---------------------------------------------------------------------------
struct AttnParams {
  const bf16_t* Qg;
  const bf16_t* Kg0; const bf16_t* Kg1; const bf16_t* Kg2;
  const bf16_t* Vt0; const bf16_t* Vt1; const bf16_t* Vt2;
  const float*  tb0; const float*  tb1; const float*  tb2;
  bf16_t* ctx;
};

__global__ __launch_bounds__(256) void attn_ms_k(AttnParams P)
{
  __shared__ __align__(16) bf16_t plds[8][16][32];
  const int lane = threadIdx.x & 31;
  const int w    = threadIdx.x >> 5;
  const int wid  = blockIdx.x * 8 + w;
  if (wid >= BB * NHH * (TT / 16)) return;
  const int b   = wid / (NHH * 36);
  const int rem = wid % (NHH * 36);
  const int h   = rem / 36;
  const int q0  = (rem % 36) * 16;
  const int n   = lane & 15;
  const int hi  = lane >> 4;

  const bf16_t* qbase = P.Qg + ((size_t)(b * TT + q0)) * DMODEL + h * HDD;
  const v16bf aq0 = ld_frag_a(qbase, DMODEL, 0);    // d = 0..31
  const v16bf aq1 = ld_frag_a(qbase, DMODEL, 32);   // d = 32..63

  v8f otot[4];
#pragma unroll
  for (int t = 0; t < 4; ++t) otot[t] = vzero8();

  const int     Sarr[3]   = {1, 2, 4};
  const int     GWarr[3]  = {24, 12, 6};
  const int     TSarr[3]  = {576, 144, 36};
  const int     TSParr[3] = {576, 160, 64};
  const int     TBSarr[3] = {47 * 47, 47 * 23, 47 * 11};
  const bf16_t* Kgs[3] = {P.Kg0, P.Kg1, P.Kg2};
  const bf16_t* Vts[3] = {P.Vt0, P.Vt1, P.Vt2};
  const float*  tbs[3] = {P.tb0, P.tb1, P.tb2};

#pragma unroll
  for (int si = 0; si < 3; ++si) {
    const int s = Sarr[si], gw = GWarr[si], Ts = TSarr[si], Tsp = TSParr[si];
    const int tw = 2 * gw - 1;
    const float cOff = (float)(s - 1) * 0.5f;
    const float* tb   = tbs[si] + (size_t)h * TBSarr[si];
    const bf16_t* Kg  = Kgs[si];
    const bf16_t* Vt  = Vts[si];

    v8f oc[4];
#pragma unroll
    for (int t = 0; t < 4; ++t) oc[t] = vzero8();
    float mrow[8], lrow[8];
#pragma unroll
    for (int v = 0; v < 8; ++v) { mrow[v] = -3.0e38f; lrow[v] = 0.0f; }

    for (int kk = 0; kk < Tsp; kk += 32) {
      // ---- scores: 16 q x 32 keys = 2 x (2 WMMAs over d) ------------------
      v8f sc[2];
#pragma unroll
      for (int j = 0; j < 2; ++j) {
        sc[j] = vzero8();
        const bf16_t* kb =
            Kg + ((size_t)(b * Tsp + kk + 16 * j)) * DMODEL + h * HDD;
        sc[j] = wmma_bf16(aq0, ld_frag_b(kb, DMODEL, 0),  sc[j]);
        sc[j] = wmma_bf16(aq1, ld_frag_b(kb, DMODEL, 32), sc[j]);
      }
      // ---- scale + rel-pos bias gather, mask pad keys ---------------------
#pragma unroll
      for (int j = 0; j < 2; ++j) {
        const int key = kk + 16 * j + n;        // this lane's key column
        if (key < Ts) {
          const int   kh = key / gw, kw = key - kh * gw;
          const float ch = (float)kh * (float)s + cOff;
          const float cw = (float)kw * (float)s + cOff;
#pragma unroll
          for (int v = 0; v < 8; ++v) {
            const int q  = q0 + v + 8 * hi;
            const int qh = q / 24, qw = q - qh * 24;
            const int rh = (int)((float)qh - ch) + 23;       // trunc-to-zero
            const int rw = (int)((float)qw - cw) + (gw - 1);
            sc[j][v] = sc[j][v] * 0.125f + tb[rh * tw + rw];
          }
        } else {
#pragma unroll
          for (int v = 0; v < 8; ++v) sc[j][v] = -3.0e38f;
        }
      }
      // ---- online softmax (rows live in 16-lane halves) -------------------
#pragma unroll
      for (int v = 0; v < 8; ++v) {
        float cm = fmaxf(sc[0][v], sc[1][v]);
        cm = fmaxf(cm, __shfl_xor(cm, 1, 32));
        cm = fmaxf(cm, __shfl_xor(cm, 2, 32));
        cm = fmaxf(cm, __shfl_xor(cm, 4, 32));
        cm = fmaxf(cm, __shfl_xor(cm, 8, 32));
        const float nm   = fmaxf(mrow[v], cm);
        const float corr = __expf(mrow[v] - nm);
        const float p0   = __expf(sc[0][v] - nm);
        const float p1   = __expf(sc[1][v] - nm);
        float rs = p0 + p1;
        rs += __shfl_xor(rs, 1, 32);
        rs += __shfl_xor(rs, 2, 32);
        rs += __shfl_xor(rs, 4, 32);
        rs += __shfl_xor(rs, 8, 32);
        lrow[v] = lrow[v] * corr + rs;
        mrow[v] = nm;
        sc[0][v] = p0; sc[1][v] = p1;
#pragma unroll
        for (int t = 0; t < 4; ++t) oc[t][v] *= corr;
      }
      // ---- re-shape P (C layout) -> A fragment via LDS --------------------
#pragma unroll
      for (int j = 0; j < 2; ++j)
#pragma unroll
        for (int v = 0; v < 8; ++v)
          plds[w][v + 8 * hi][16 * j + n] = (bf16_t)sc[j][v];
      asm volatile("s_wait_dscnt 0" ::: "memory");
      const v16bf pa = ld_frag_a(&plds[w][0][0], 32, 0);
      // ---- PV: 4 WMMAs over d tiles ---------------------------------------
#pragma unroll
      for (int t = 0; t < 4; ++t) {
        const bf16_t* vb = Vt + ((size_t)((b * NHH + h) * HDD + 16 * t)) * Tsp;
        oc[t] = wmma_bf16(pa, ld_frag_b(vb, Tsp, kk), oc[t]);
      }
    }
    // ---- fold this scale into the accumulator -----------------------------
#pragma unroll
    for (int v = 0; v < 8; ++v) {
      const float inv = 1.0f / lrow[v];
#pragma unroll
      for (int t = 0; t < 4; ++t) otot[t][v] += oc[t][v] * inv;
    }
  }
  // ---- store ctx (bf16, row-major (B*T, 768)) -----------------------------
#pragma unroll
  for (int t = 0; t < 4; ++t)
#pragma unroll
    for (int v = 0; v < 8; ++v) {
      const int row = b * TT + q0 + v + 8 * hi;
      const int col = h * HDD + 16 * t + n;
      P.ctx[(size_t)row * DMODEL + col] = (bf16_t)(otot[t][v] * (1.0f / 3.0f));
    }
}

// ---------------------------------------------------------------------------
// Host-side orchestration
// ---------------------------------------------------------------------------
static inline int cdiv(int a, int b) { return (a + b - 1) / b; }

extern "C" void kernel_launch(void* const* d_in, const int* in_sizes, int n_in,
                              void* d_out, int out_size, void* d_ws, size_t ws_size,
                              hipStream_t stream)
{
  (void)in_sizes; (void)n_in; (void)out_size; (void)ws_size;

  const float* x    = (const float*)d_in[0];
  const float* wq   = (const float*)d_in[1];
  const float* wdkv = (const float*)d_in[2];
  const float* wuk[3] = {(const float*)d_in[3], (const float*)d_in[4],
                         (const float*)d_in[5]};
  const float* wuv[3] = {(const float*)d_in[6], (const float*)d_in[7],
                         (const float*)d_in[8]};
  const float* wout = (const float*)d_in[9];
  const float* bout = (const float*)d_in[10];
  const float* tbl[3] = {(const float*)d_in[11], (const float*)d_in[12],
                         (const float*)d_in[13]};
  float* out = (float*)d_out;

  const int BT = BB * TT;                       // 9216
  const int Sarr[3]   = {1, 2, 4};
  const int GWarr[3]  = {24, 12, 6};
  const int TSarr[3]  = {576, 144, 36};
  const int TSParr[3] = {576, 160, 64};         // padded keys (mult of 32)

  // ---- workspace bump allocator (256B aligned) ----------------------------
  char*  base = (char*)d_ws;
  size_t off  = 0;
  auto alloc = [&](size_t bytes) -> void* {
    void* p = (void*)(base + off);
    off += (bytes + 255) & ~(size_t)255;
    return p;
  };

  bf16_t* xb     = (bf16_t*)alloc((size_t)BT * DMODEL * 2);
  bf16_t* wqT    = (bf16_t*)alloc((size_t)DMODEL * DMODEL * 2);
  bf16_t* wdkvT  = (bf16_t*)alloc((size_t)LDIM * DMODEL * 2);
  bf16_t* woutT  = (bf16_t*)alloc((size_t)DMODEL * DMODEL * 2);
  bf16_t* wukT[3], *wuvT[3], *lp[3], *Kg[3], *Vg[3], *Vt[3];
  for (int i = 0; i < 3; ++i) {
    wukT[i] = (bf16_t*)alloc((size_t)DMODEL * LDIM * 2);
    wuvT[i] = (bf16_t*)alloc((size_t)DMODEL * LDIM * 2);
  }
  bf16_t* Qg  = (bf16_t*)alloc((size_t)BT * DMODEL * 2);
  float*  lat = (float*)alloc((size_t)BT * LDIM * 4);
  for (int i = 0; i < 3; ++i) {
    lp[i] = (bf16_t*)alloc((size_t)BB * TSParr[i] * LDIM * 2);
    Kg[i] = (bf16_t*)alloc((size_t)BB * TSParr[i] * DMODEL * 2);
    Vg[i] = (bf16_t*)alloc((size_t)BB * TSParr[i] * DMODEL * 2);
    Vt[i] = (bf16_t*)alloc((size_t)BB * NHH * HDD * TSParr[i] * 2);
  }
  bf16_t* ctx = (bf16_t*)alloc((size_t)BT * DMODEL * 2);

  auto gemm = [&](const bf16_t* A, const bf16_t* Bt, bf16_t* Cb, float* Cf,
                  const float* bias, int M, int N, int K) {
    int waves  = (M >> 6) * (N >> 5);
    int blocks = cdiv(waves, 8);
    gemm_bf16_wmma<<<blocks, 256, 0, stream>>>(A, Bt, Cb, Cf, bias, M, N, K);
  };

  // 1) precision / layout transforms
  {
    int nx = BT * DMODEL;
    f32_to_bf16_k<<<cdiv(nx, 256), 256, 0, stream>>>(x, xb, nx);
    transpose_to_bf16_k<<<cdiv(DMODEL * DMODEL, 256), 256, 0, stream>>>(
        wq, wqT, DMODEL, DMODEL);
    transpose_to_bf16_k<<<cdiv(DMODEL * LDIM, 256), 256, 0, stream>>>(
        wdkv, wdkvT, DMODEL, LDIM);
    transpose_to_bf16_k<<<cdiv(DMODEL * DMODEL, 256), 256, 0, stream>>>(
        wout, woutT, DMODEL, DMODEL);
    for (int i = 0; i < 3; ++i) {
      transpose_to_bf16_k<<<cdiv(LDIM * DMODEL, 256), 256, 0, stream>>>(
          wuk[i], wukT[i], LDIM, DMODEL);
      transpose_to_bf16_k<<<cdiv(LDIM * DMODEL, 256), 256, 0, stream>>>(
          wuv[i], wuvT[i], LDIM, DMODEL);
    }
  }

  // 2) Q = x @ wq (bf16 out), lat = x @ wdkv (f32 out)
  gemm(xb, wqT,   Qg,      nullptr, nullptr, BT, DMODEL, DMODEL);
  gemm(xb, wdkvT, nullptr, lat,     nullptr, BT, LDIM,   DMODEL);

  // 3) per-scale pooling, K/V up-projection, V transpose
  for (int i = 0; i < 3; ++i) {
    int np = BB * TSParr[i] * LDIM;
    pool_lat_k<<<cdiv(np, 256), 256, 0, stream>>>(
        lat, lp[i], Sarr[i], GWarr[i], TSarr[i], TSParr[i]);
    gemm(lp[i], wukT[i], Kg[i], nullptr, nullptr, BB * TSParr[i], DMODEL, LDIM);
    gemm(lp[i], wuvT[i], Vg[i], nullptr, nullptr, BB * TSParr[i], DMODEL, LDIM);
    int nv = BB * NHH * HDD * TSParr[i];
    v_transpose_k<<<cdiv(nv, 256), 256, 0, stream>>>(Vg[i], Vt[i], TSParr[i]);
  }

  // 4) multi-scale flash attention
  {
    AttnParams P;
    P.Qg = Qg;
    P.Kg0 = Kg[0]; P.Kg1 = Kg[1]; P.Kg2 = Kg[2];
    P.Vt0 = Vt[0]; P.Vt1 = Vt[1]; P.Vt2 = Vt[2];
    P.tb0 = tbl[0]; P.tb1 = tbl[1]; P.tb2 = tbl[2];
    P.ctx = ctx;
    int waves  = BB * NHH * (TT / 16);          // 6912
    int blocks = cdiv(waves, 8);                // 864
    attn_ms_k<<<blocks, 256, 0, stream>>>(P);
  }

  // 5) out = ctx @ wout + bout (f32)
  gemm(ctx, woutT, nullptr, out, bout, BT, DMODEL, DMODEL);
}
